// EdgeConv_11416023073361
// MI455X (gfx1250) — matrix operand order repaired and checked by reading, not compiled, not run on hardware
//
#include <hip/hip_runtime.h>
#include <hip/hip_bf16.h>

// ---------------- problem constants ----------------
#define N_NODES 100000
#define N_EDGES 1600000
#define IN_DIM  128
#define OUT_DIM 128
#define EDGE_DIM 32
#define HID 256                       // msg hidden width
#define MSG_K (IN_DIM + EDGE_DIM)     // 160
#define UPD_K (IN_DIM + OUT_DIM)      // 256
#define LN_EPS 1e-5f

typedef __attribute__((ext_vector_type(16))) _Float16 v16h;
typedef __attribute__((ext_vector_type(8)))  _Float16 v8h;
typedef __attribute__((ext_vector_type(4)))  _Float16 v4h;
typedef __attribute__((ext_vector_type(8)))  float    v8f;
typedef __attribute__((ext_vector_type(4)))  float    v4f;

// padded LDS strides (elements)
#define A_STR   168   // 160 padded
#define H_STR   264   // 256 padded
#define U_STR   136   // 128 padded

// ---- edge-kernel LDS layout (bytes) : only activations now ----
#define EAF_OFF  0                    // 16*168 f16 = 5376
#define EH32_OFF 5376                 // 16*264 f32 = 16896
#define EHH_OFF  22272                // 16*264 f16 = 8448
#define EB1_OFF  30720                // 256 f32
#define EG1_OFF  31744
#define EBE1_OFF 32768
#define EB2_OFF  33792                // 128 f32
#define EROW_OFF 34304                // 16 int
#define ECOL_OFF 34368                // 16 int
#define EDGE_LDS 34432

// ---- node-kernel LDS layout ----
#define NAF_OFF  0                    // 16*264 f16 = 8448
#define NU32_OFF 8448                 // 16*136 f32 = 8704
#define NBU_OFF  17152
#define NGU_OFF  17664
#define NBEU_OFF 18176
#define NODE_LDS 18688

// ---- workspace layout (bytes) ----
#define AGG_OFF  0                               // N*128 f32 = 51,200,000
#define W1T_OFF  51200000                        // 256*160 f16 = 81920
#define W2T_OFF  (W1T_OFF + 81920)               // 128*256 f16 = 65536
#define WUT_OFF  (W2T_OFF + 65536)               // 128*256 f16 = 65536
#define WS_MIN   (WUT_OFF + 65536)               // = 51,412,992
#define XH_OFF   WS_MIN                          // N*128 f16 = 25,600,000 (256B aligned)
#define WS_FULL  (XH_OFF + 25600000)

// ------------------------------------------------------------------
// A fragment: 16(M) x 32(K), f16, from LDS (row-major, padded stride).
__device__ __forceinline__ v16h ldsA16x32(const _Float16* base, int stride, int k0) {
    const int lane = threadIdx.x & 31;
    const int m  = lane & 15;
    const int kh = (lane & 16) ? 8 : 0;
    const _Float16* p = base + m * stride + k0 + kh;
    v8h lo = *(const v8h*)(p);
    v8h hi = *(const v8h*)(p + 16);
    v16h r;
#pragma unroll
    for (int i = 0; i < 8; ++i) { r[i] = lo[i]; r[i + 8] = hi[i]; }
    return r;
}

// B fragment: 32(K) x 16(N), from GLOBAL transposed weights BT[n][k] (L2-hot).
__device__ __forceinline__ v16h glbB32x16(const _Float16* baseT, int stride, int n0, int k0) {
    const int lane = threadIdx.x & 31;
    const int n  = lane & 15;
    const int ko = (lane & 16) ? 8 : 0;
    const _Float16* p = baseT + (n0 + n) * stride + k0 + ko;
    v8h lo = *(const v8h*)(p);
    v8h hi = *(const v8h*)(p + 16);
    v16h r;
#pragma unroll
    for (int i = 0; i < 8; ++i) { r[i] = lo[i]; r[i + 8] = hi[i]; }
    return r;
}

__device__ __forceinline__ v8f wmma_f16(v16h a, v16h b, v8f c) {
    return __builtin_amdgcn_wmma_f32_16x16x32_f16(false, a, false, b, (short)0, c, false, false);
}

__device__ __forceinline__ float gelu_exact(float x) {
    return 0.5f * x * (1.0f + erff(x * 0.70710678118654752f));
}

// 16-byte async global->LDS copy (ASYNCcnt-tracked, no VGPR round trip).
__device__ __forceinline__ void async_copy_b128(void* ldsPtr, const void* gPtr) {
    unsigned lds_addr = (unsigned)(uintptr_t)ldsPtr;   // low 32 bits of generic = LDS offset
    asm volatile("global_load_async_to_lds_b128 %0, %1, off"
                 :: "v"(lds_addr), "v"(gPtr) : "memory");
}
__device__ __forceinline__ void wait_async0() {
    asm volatile("s_wait_asynccnt 0x0" ::: "memory");
}

__device__ __forceinline__ v4h cvt4(v4f v) {
    v4h h;
#pragma unroll
    for (int i = 0; i < 4; ++i) h[i] = (_Float16)v[i];
    return h;
}

// ------------------------------------------------------------------
// Prep: zero agg accumulator; transpose+convert weights to f16; x -> f16.
__global__ void edgeconv_prep_kernel(
    const float* __restrict__ x,
    const float* __restrict__ W1, const float* __restrict__ W2,
    const float* __restrict__ Wu,
    float* __restrict__ agg, _Float16* __restrict__ w1t,
    _Float16* __restrict__ w2t, _Float16* __restrict__ wut,
    _Float16* __restrict__ xh /* may be null */)
{
    const int stride = gridDim.x * blockDim.x;
    const int tid0 = blockIdx.x * blockDim.x + threadIdx.x;
    for (int i = tid0; i < N_NODES * OUT_DIM; i += stride) agg[i] = 0.0f;
    for (int i = tid0; i < HID * MSG_K; i += stride) {            // w1t[n][k]=W1[k][n]
        int n = i / MSG_K, k = i - n * MSG_K;
        w1t[n * MSG_K + k] = (_Float16)W1[k * HID + n];
    }
    for (int i = tid0; i < OUT_DIM * HID; i += stride) {          // w2t[n][k]=W2[k][n]
        int n = i >> 8, k = i & 255;
        w2t[n * HID + k] = (_Float16)W2[k * OUT_DIM + n];
    }
    for (int i = tid0; i < OUT_DIM * UPD_K; i += stride) {        // wut[n][k]=Wu[k][n]
        int n = i >> 8, k = i & 255;
        wut[n * UPD_K + k] = (_Float16)Wu[k * OUT_DIM + n];
    }
    if (xh) {
        for (int i = tid0; i < N_NODES * IN_DIM; i += stride)
            xh[i] = (_Float16)x[i];
    }
}

// ------------------------------------------------------------------
// Edge kernel: 16-edge tiles; WMMA msg-MLP; f32 atomic scatter into agg.
// All weight B-fragments held in VGPRs for the whole persistent loop.
__global__ __launch_bounds__(256) void edgeconv_edge_kernel(
    const float* __restrict__ x, const long long* __restrict__ edge_index,
    const float* __restrict__ ea, const _Float16* __restrict__ xh /* may be null */,
    const _Float16* __restrict__ w1t, const _Float16* __restrict__ w2t,
    const float* __restrict__ b1, const float* __restrict__ g1,
    const float* __restrict__ be1, const float* __restrict__ b2,
    float* __restrict__ agg)
{
    extern __shared__ char smem[];
    _Float16* Af  = (_Float16*)(smem + EAF_OFF);
    float*    h32 = (float*)(smem + EH32_OFF);
    _Float16* Hh  = (_Float16*)(smem + EHH_OFF);
    float*    b1s = (float*)(smem + EB1_OFF);
    float*    g1s = (float*)(smem + EG1_OFF);
    float*    be1s= (float*)(smem + EBE1_OFF);
    float*    b2s = (float*)(smem + EB2_OFF);
    int*      rowb= (int*)(smem + EROW_OFF);
    int*      colb= (int*)(smem + ECOL_OFF);

    const int tid  = threadIdx.x;
    const int wave = tid >> 5;
    const int lane = tid & 31;

    if (tid < HID) { b1s[tid] = b1[tid]; g1s[tid] = g1[tid]; be1s[tid] = be1[tid]; }
    if (tid < OUT_DIM) b2s[tid] = b2[tid];

    // ---- hoist all weight fragments into VGPRs (loop-invariant, L2-hot) ----
    const int N1 = wave * 32;   // GEMM1 column base (hidden)
    const int N2 = wave * 16;   // GEMM2 column base (output)
    v16h B1a[5], B1b[5], B2r[8];
#pragma unroll
    for (int ks = 0; ks < 5; ++ks) {
        B1a[ks] = glbB32x16(w1t, MSG_K, N1,      ks * 32);
        B1b[ks] = glbB32x16(w1t, MSG_K, N1 + 16, ks * 32);
    }
#pragma unroll
    for (int ks = 0; ks < 8; ++ks)
        B2r[ks] = glbB32x16(w2t, HID, N2, ks * 32);
    __syncthreads();

    const int numTiles = N_EDGES / 16;
    for (int t = blockIdx.x; t < numTiles; t += gridDim.x) {
        const int e0 = t * 16;
        if (tid < 16) {
            rowb[tid] = (int)edge_index[e0 + tid];
            colb[tid] = (int)edge_index[N_EDGES + e0 + tid];
        }
        __syncthreads();

        // ---- stage A panel [16][160] f16 = [x[row] | edge_attr] ----
        if (xh) {
            // async 16B copies from pre-converted f16 x (gfx1250 async-LDS path)
            const int m = tid >> 4, seg = tid & 15;            // 256 threads = 256 copies
            async_copy_b128(&Af[m * A_STR + seg * 8],
                            xh + (size_t)rowb[m] * IN_DIM + seg * 8);
        } else {
            for (int i = tid; i < 16 * 32; i += 256) {         // float4 convert path
                int m = i >> 5, c4 = i & 31;
                v4f v = *(const v4f*)(x + (size_t)rowb[m] * IN_DIM + c4 * 4);
                *(v4h*)(&Af[m * A_STR + c4 * 4]) = cvt4(v);
            }
        }
        if (tid < 16 * 8) {                                    // edge_attr: 16 rows x 8 float4
            int m = tid >> 3, c4 = tid & 7;
            v4f v = *(const v4f*)(ea + (size_t)(e0 + m) * EDGE_DIM + c4 * 4);
            *(v4h*)(&Af[m * A_STR + IN_DIM + c4 * 4]) = cvt4(v);
        }
        wait_async0();
        __syncthreads();

        // ---- GEMM1: H[16][256] = A @ W1 ; wave owns hidden cols [32w,32w+32) ----
        {
            v8f acc0 = {}; v8f acc1 = {};
#pragma unroll
            for (int ks = 0; ks < 5; ++ks) {
                v16h a = ldsA16x32(Af, A_STR, ks * 32);
                acc0 = wmma_f16(a, B1a[ks], acc0);
                acc1 = wmma_f16(a, B1b[ks], acc1);
            }
            const int n  = lane & 15;
            const int mb = (lane >> 4) * 8;
            float bia0 = b1s[N1 + n], bia1 = b1s[N1 + 16 + n];
#pragma unroll
            for (int v = 0; v < 8; ++v) {
                h32[(mb + v) * H_STR + N1 + n]      = acc0[v] + bia0;
                h32[(mb + v) * H_STR + N1 + 16 + n] = acc1[v] + bia1;
            }
        }
        __syncthreads();

        // ---- LayerNorm + exact GELU (256 wide); wave w: rows 2w, 2w+1 ----
        {
            const int row   = 2 * wave + (lane >> 4);
            const int cbase = (lane & 15) * 16;
            float s = 0.f, s2 = 0.f;
#pragma unroll
            for (int j = 0; j < 16; ++j) {
                float v = h32[row * H_STR + cbase + j];
                s += v; s2 += v * v;
            }
#pragma unroll
            for (int off = 1; off < 16; off <<= 1) {
                s  += __shfl_xor(s,  off, 32);
                s2 += __shfl_xor(s2, off, 32);
            }
            const float mu   = s * (1.0f / HID);
            const float var  = s2 * (1.0f / HID) - mu * mu;
            const float rstd = rsqrtf(var + LN_EPS);
#pragma unroll
            for (int j = 0; j < 16; ++j) {
                int c = cbase + j;
                float v = (h32[row * H_STR + c] - mu) * rstd * g1s[c] + be1s[c];
                Hh[row * H_STR + c] = (_Float16)gelu_exact(v);
            }
        }
        __syncthreads();

        // ---- GEMM2: msg[16][128] = H @ W2 ; wave owns out cols [16w,16w+16) ----
        {
            v8f acc = {};
#pragma unroll
            for (int ks = 0; ks < 8; ++ks) {
                v16h a = ldsA16x32(Hh, H_STR, ks * 32);
                acc = wmma_f16(a, B2r[ks], acc);
            }
            const int n  = lane & 15;
            const int mb = (lane >> 4) * 8;
            const float bia = b2s[N2 + n];
            const unsigned cbase = (unsigned)(N2 + n);
#pragma unroll
            for (int v = 0; v < 8; ++v) {
                unsigned off = (unsigned)colb[mb + v] * (unsigned)OUT_DIM + cbase;
                atomicAdd(&agg[off], acc[v] + bia);
            }
        }
        __syncthreads();   // protect rowb/colb/Af/Hh for next tile
    }
}

// ------------------------------------------------------------------
// Node kernel: u = GELU(LN(concat(x,agg) @ Wu + bu)) + x
__global__ __launch_bounds__(256) void edgeconv_node_kernel(
    const float* __restrict__ x, const _Float16* __restrict__ xh /* may be null */,
    const float* __restrict__ agg, const _Float16* __restrict__ wut,
    const float* __restrict__ bu, const float* __restrict__ gu,
    const float* __restrict__ beu, float* __restrict__ out)
{
    extern __shared__ char smem[];
    _Float16* Af  = (_Float16*)(smem + NAF_OFF);
    float*    u32 = (float*)(smem + NU32_OFF);
    float*    bus = (float*)(smem + NBU_OFF);
    float*    gus = (float*)(smem + NGU_OFF);
    float*    beus= (float*)(smem + NBEU_OFF);

    const int tid  = threadIdx.x;
    const int wave = tid >> 5;
    const int lane = tid & 31;

    if (tid < OUT_DIM) { bus[tid] = bu[tid]; gus[tid] = gu[tid]; beus[tid] = beu[tid]; }

    const int N0 = wave * 16;
    v16h Bu[8];
#pragma unroll
    for (int ks = 0; ks < 8; ++ks)
        Bu[ks] = glbB32x16(wut, UPD_K, N0, ks * 32);
    __syncthreads();

    const int numTiles = N_NODES / 16;
    for (int t = blockIdx.x; t < numTiles; t += gridDim.x) {
        const int n0 = t * 16;

        // stage A panel [16][256] = [x | agg] in f16
        if (xh) {
            const int m = tid >> 4, seg = tid & 15;
            async_copy_b128(&Af[m * H_STR + seg * 8],
                            xh + (size_t)(n0 + m) * IN_DIM + seg * 8);
        } else {
            for (int i = tid; i < 16 * 32; i += 256) {
                int m = i >> 5, c4 = i & 31;
                v4f v = *(const v4f*)(x + (size_t)(n0 + m) * IN_DIM + c4 * 4);
                *(v4h*)(&Af[m * H_STR + c4 * 4]) = cvt4(v);
            }
        }
        for (int i = tid; i < 16 * 32; i += 256) {             // agg half (f32 -> f16)
            int m = i >> 5, c4 = i & 31;
            v4f v = *(const v4f*)(agg + (size_t)(n0 + m) * OUT_DIM + c4 * 4);
            *(v4h*)(&Af[m * H_STR + IN_DIM + c4 * 4]) = cvt4(v);
        }
        wait_async0();
        __syncthreads();

        // GEMM: u[16][128] ; wave owns cols [16w,16w+16)
        {
            v8f acc = {};
#pragma unroll
            for (int ks = 0; ks < 8; ++ks) {
                v16h a = ldsA16x32(Af, H_STR, ks * 32);
                acc = wmma_f16(a, Bu[ks], acc);
            }
            const int n  = lane & 15;
            const int mb = (lane >> 4) * 8;
            const float bia = bus[N0 + n];
#pragma unroll
            for (int v = 0; v < 8; ++v)
                u32[(mb + v) * U_STR + N0 + n] = acc[v] + bia;
        }
        __syncthreads();

        // LN + GELU + residual ; wave w: rows 2w,2w+1 ; lane covers 8 cols
        {
            const int row   = 2 * wave + (lane >> 4);
            const int cbase = (lane & 15) * 8;
            float s = 0.f, s2 = 0.f;
#pragma unroll
            for (int j = 0; j < 8; ++j) {
                float v = u32[row * U_STR + cbase + j];
                s += v; s2 += v * v;
            }
#pragma unroll
            for (int off = 1; off < 16; off <<= 1) {
                s  += __shfl_xor(s,  off, 32);
                s2 += __shfl_xor(s2, off, 32);
            }
            const float mu   = s * (1.0f / OUT_DIM);
            const float var  = s2 * (1.0f / OUT_DIM) - mu * mu;
            const float rstd = rsqrtf(var + LN_EPS);
            const size_t gbase = (size_t)(n0 + row) * OUT_DIM;
#pragma unroll
            for (int j = 0; j < 8; ++j) {
                int c = cbase + j;
                float v = (u32[row * U_STR + c] - mu) * rstd * gus[c] + beus[c];
                out[gbase + c] = gelu_exact(v) + x[gbase + c];
            }
        }
        __syncthreads();
    }
}

// ------------------------------------------------------------------
extern "C" void kernel_launch(void* const* d_in, const int* in_sizes, int n_in,
                              void* d_out, int out_size, void* d_ws, size_t ws_size,
                              hipStream_t stream) {
    const float*     x   = (const float*)d_in[0];
    const long long* ei  = (const long long*)d_in[1];
    const float*     ea  = (const float*)d_in[2];
    const float*     W1  = (const float*)d_in[3];
    const float*     b1  = (const float*)d_in[4];
    const float*     g1  = (const float*)d_in[5];
    const float*     be1 = (const float*)d_in[6];
    const float*     W2  = (const float*)d_in[7];
    const float*     b2  = (const float*)d_in[8];
    const float*     Wu  = (const float*)d_in[9];
    const float*     bu  = (const float*)d_in[10];
    const float*     gu  = (const float*)d_in[11];
    const float*     beu = (const float*)d_in[12];
    float* out = (float*)d_out;

    char* ws = (char*)d_ws;
    float*     agg = (float*)(ws + AGG_OFF);
    _Float16*  w1t = (_Float16*)(ws + W1T_OFF);
    _Float16*  w2t = (_Float16*)(ws + W2T_OFF);
    _Float16*  wut = (_Float16*)(ws + WUT_OFF);
    _Float16*  xh  = (ws_size >= (size_t)WS_FULL) ? (_Float16*)(ws + XH_OFF) : (_Float16*)0;

    (void)in_sizes; (void)n_in; (void)out_size;

    edgeconv_prep_kernel<<<2048, 256, 0, stream>>>(x, W1, W2, Wu, agg, w1t, w2t, wut, xh);
    edgeconv_edge_kernel<<<4096, 256, EDGE_LDS, stream>>>(
        x, ei, ea, xh, w1t, w2t, b1, g1, be1, b2, agg);
    edgeconv_node_kernel<<<2048, 256, NODE_LDS, stream>>>(
        x, xh, agg, wut, bu, gu, beu, out);
}
